// SparseAttention_40707700032208
// MI455X (gfx1250) — matrix operand order repaired and checked by reading, not compiled, not run on hardware
//
#include <hip/hip_runtime.h>
#include <hip/hip_bf16.h>

// CDNA5 / gfx1250 fused sparse (1.5-entmax) attention.
// B=8, S=2048, D=128, fp32 in/out. One workgroup = 8 wave32s = one 16-query tile.

typedef __attribute__((ext_vector_type(16))) __bf16 v16bf;
typedef __attribute__((ext_vector_type(8)))  float  v8f;
typedef __attribute__((ext_vector_type(2)))  float  f32x2;

#define SEQ   2048
#define DHEAD 128
#define NITER 50
// 2048^(1-1.5) = 1/sqrt(2048)
#define HI_OFF 0.02209708691207961f

// ---- f32 pair -> packed bf16 dword ------------------------------------------
#if __has_builtin(__builtin_amdgcn_cvt_pk_bf16_f32)
static __device__ __forceinline__ unsigned pk2u(float a, float b) {
    auto r = __builtin_amdgcn_cvt_pk_bf16_f32(a, b);
    return __builtin_bit_cast(unsigned, r);
}
#else
// v_perm_b32: result = {hi16(b+bias), hi16(a+bias)} in 3 VALU ops.
static __device__ __forceinline__ unsigned pk2u(float a, float b) {
    unsigned ua = __builtin_bit_cast(unsigned, a) + 0x8000u;
    unsigned ub = __builtin_bit_cast(unsigned, b) + 0x8000u;
    return __builtin_amdgcn_perm(ub, ua, 0x07060302u);
}
#endif

union FragU {
    v16bf    v;
    unsigned u32[8];
};

__global__ __launch_bounds__(256)
void entmax_attn_kernel(const float* __restrict__ Q,
                        const float* __restrict__ K,
                        const float* __restrict__ V,
                        float* __restrict__ O) {
    // 128 KB: 16 rows x 2048 f32 scores; later reused as 16 x 2048 bf16 weights.
    __shared__ float Slds[16 * SEQ];
    unsigned* Wlds32 = reinterpret_cast<unsigned*>(Slds);

    const int b     = blockIdx.x >> 7;          // 128 query tiles per batch
    const int qBase = (blockIdx.x & 127) << 4;  // 16 queries per tile
    const int w     = threadIdx.x >> 5;         // wave id 0..7
    const int lane  = threadIdx.x & 31;
    const int lm    = lane & 15;
    const int hi16  = (lane >> 4) & 1;          // upper half-wave flag
    const int hi8   = hi16 ? 8 : 0;
    const int off16 = hi16 ? 16 : 0;

    // ---------------- Phase 1: x = 0.5 * Q*K^T (bf16 WMMA, f32 accum) --------
    // Fold the (alpha-1)=0.5 scale into Q's bf16 conversion (done once).
    FragU qa[4];
    {
        const float* qrow = Q + ((size_t)(b * SEQ + qBase + lm)) * DHEAD;
#pragma unroll
        for (int kb = 0; kb < 4; ++kb) {
#pragma unroll
            for (int j = 0; j < 8; ++j) {
                int e  = 2 * j;   // pairs have consecutive k per ISA A layout
                int k0 = 32 * kb + (e & 7) + ((e & 8) ? 16 : 0) + hi8;
                qa[kb].u32[j] = pk2u(0.5f * qrow[k0], 0.5f * qrow[k0 + 1]);
            }
        }
    }

    for (int t = w; t < SEQ / 16; t += 8) {
        const int nBase = t << 4;
        const float* krow = K + ((size_t)(b * SEQ + nBase + lm)) * DHEAD;
        if (t + 8 < SEQ / 16)
            __builtin_prefetch(krow + 128 * DHEAD, 0, 1);   // global_prefetch_b8

        v8f acc = {0.f, 0.f, 0.f, 0.f, 0.f, 0.f, 0.f, 0.f};
#pragma unroll
        for (int kb = 0; kb < 4; ++kb) {
            FragU bf;   // B fragment: k = (e&7) + 16*(e>=8) + 8*hi16, n = lm (K row contiguous)
#pragma unroll
            for (int j = 0; j < 8; ++j) {
                int k0 = 32 * kb + ((2 * j) & 7) + (((2 * j) & 8) ? 16 : 0) + hi8;
                bf.u32[j] = pk2u(krow[k0], krow[k0 + 1]);
            }
            acc = __builtin_amdgcn_wmma_f32_16x16x32_bf16(
                      false, qa[kb].v, false, bf.v, (short)0, acc, false, false);
        }
        // C/D layout: m = j + 8*hi16
#pragma unroll
        for (int j = 0; j < 8; ++j)
            Slds[(j + hi8) * SEQ + nBase + lm] = acc[j];
    }
    __syncthreads();

    // ---------------- Phase 2: entmax bisection, register-resident, packed ---
    // Wave w owns rows 2w, 2w+1. Lane owns contiguous column pairs
    // cols = 2*lane + 64*i  ->  b64 LDS loads, b32 weight stores.
    const int r0 = 2 * w, r1 = 2 * w + 1;
    f32x2 x0v[32], x1v[32];
    {
        const f32x2* p0 = reinterpret_cast<const f32x2*>(&Slds[r0 * SEQ + 2 * lane]);
        const f32x2* p1 = reinterpret_cast<const f32x2*>(&Slds[r1 * SEQ + 2 * lane]);
#pragma unroll
        for (int i = 0; i < 32; ++i) { x0v[i] = p0[32 * i]; x1v[i] = p1[32 * i]; }
    }
    __syncthreads();   // everyone captured x before LDS is reused for weights

    f32x2 m0 = x0v[0], m1 = x1v[0];
#pragma unroll
    for (int i = 1; i < 32; ++i) {
        m0 = __builtin_elementwise_max(m0, x0v[i]);
        m1 = __builtin_elementwise_max(m1, x1v[i]);
    }
    float mx0 = fmaxf(m0.x, m0.y), mx1 = fmaxf(m1.x, m1.y);
#pragma unroll
    for (int m = 16; m >= 1; m >>= 1) {
        mx0 = fmaxf(mx0, __shfl_xor(mx0, m, 32));
        mx1 = fmaxf(mx1, __shfl_xor(mx1, m, 32));
    }

    float lo0 = mx0 - 1.0f, hi0 = mx0 - HI_OFF;
    float lo1 = mx1 - 1.0f, hi1 = mx1 - HI_OFF;
    const f32x2 zero2 = {0.f, 0.f};

#pragma unroll 1
    for (int it = 0; it < NITER; ++it) {
        float mid0 = 0.5f * (lo0 + hi0), mid1 = 0.5f * (lo1 + hi1);
        f32x2 mid0v; mid0v.x = mid0; mid0v.y = mid0;
        f32x2 mid1v; mid1v.x = mid1; mid1v.y = mid1;
        f32x2 s0v = zero2, s1v = zero2;
#pragma unroll
        for (int i = 0; i < 32; ++i) {
            f32x2 t0 = __builtin_elementwise_max(x0v[i] - mid0v, zero2);
            s0v = __builtin_elementwise_fma(t0, t0, s0v);   // v_pk_fma_f32
            f32x2 t1 = __builtin_elementwise_max(x1v[i] - mid1v, zero2);
            s1v = __builtin_elementwise_fma(t1, t1, s1v);
        }
        float s0 = s0v.x + s0v.y, s1 = s1v.x + s1v.y;
#pragma unroll
        for (int m = 16; m >= 1; m >>= 1) {
            s0 += __shfl_xor(s0, m, 32);
            s1 += __shfl_xor(s1, m, 32);
        }
        if (s0 >= 1.0f) lo0 = mid0; else hi0 = mid0;   // uniform across wave
        if (s1 >= 1.0f) lo1 = mid1; else hi1 = mid1;
    }

    const float tau0 = 0.5f * (lo0 + hi0), tau1 = 0.5f * (lo1 + hi1);
    f32x2 tau0v; tau0v.x = tau0; tau0v.y = tau0;
    f32x2 tau1v; tau1v.x = tau1; tau1v.y = tau1;
    f32x2 s0v = zero2, s1v = zero2;
#pragma unroll
    for (int i = 0; i < 32; ++i) {
        f32x2 t0 = __builtin_elementwise_max(x0v[i] - tau0v, zero2);
        f32x2 p0 = t0 * t0; x0v[i] = p0; s0v = s0v + p0;
        f32x2 t1 = __builtin_elementwise_max(x1v[i] - tau1v, zero2);
        f32x2 p1 = t1 * t1; x1v[i] = p1; s1v = s1v + p1;
    }
    float s0 = s0v.x + s0v.y, s1 = s1v.x + s1v.y;
#pragma unroll
    for (int m = 16; m >= 1; m >>= 1) {
        s0 += __shfl_xor(s0, m, 32);
        s1 += __shfl_xor(s1, m, 32);
    }
    const float inv0 = 1.0f / s0, inv1 = 1.0f / s1;

    // normalized weights -> LDS bf16 (row-major 16 x 2048); one b32 store/pair
#pragma unroll
    for (int i = 0; i < 32; ++i) {
        Wlds32[(r0 * SEQ + 2 * lane + 64 * i) >> 1] =
            pk2u(x0v[i].x * inv0, x0v[i].y * inv0);
        Wlds32[(r1 * SEQ + 2 * lane + 64 * i) >> 1] =
            pk2u(x1v[i].x * inv1, x1v[i].y * inv1);
    }
    __syncthreads();

    // ---------------- Phase 3: out = W * V (bf16 WMMA, f32 accum) ------------
    // Wave w owns output columns dBase..dBase+15 (8 waves * 16 = 128 = DHEAD).
    const int dBase = w << 4;
    v8f oacc = {0.f, 0.f, 0.f, 0.f, 0.f, 0.f, 0.f, 0.f};
    for (int t = 0; t < SEQ / 32; ++t) {
        const int kBase = t << 5;
        FragU af, bfr;
#pragma unroll
        for (int j = 0; j < 8; ++j) {
            int e  = 2 * j;
            int k0 = kBase + e + off16;  // even -> dword
            af.u32[j] = Wlds32[(lm * SEQ + k0) >> 1];
        }
        const float* vb = V + ((size_t)(b * SEQ + kBase + off16)) * DHEAD
                            + dBase + lm;
        if (t + 1 < SEQ / 32)
            __builtin_prefetch(vb + 32 * DHEAD, 0, 1);
#pragma unroll
        for (int j = 0; j < 8; ++j)
            bfr.u32[j] = pk2u(vb[(size_t)(2 * j) * DHEAD],
                              vb[(size_t)(2 * j + 1) * DHEAD]);
        oacc = __builtin_amdgcn_wmma_f32_16x16x32_bf16(
                   false, af.v, false, bfr.v, (short)0, oacc, false, false);
    }
#pragma unroll
    for (int j = 0; j < 8; ++j) {
        int m = j + hi8;
        O[((size_t)(b * SEQ + qBase + m)) * DHEAD + dBase + lm] = oacc[j];
    }
}

extern "C" void kernel_launch(void* const* d_in, const int* in_sizes, int n_in,
                              void* d_out, int out_size, void* d_ws, size_t ws_size,
                              hipStream_t stream) {
    (void)in_sizes; (void)n_in; (void)d_ws; (void)ws_size; (void)out_size;
    const float* Q = (const float*)d_in[0];
    const float* K = (const float*)d_in[1];
    const float* V = (const float*)d_in[2];
    float* O = (float*)d_out;
    // 8 batches * 128 query tiles = 1024 workgroups of 256 threads (8 wave32s)
    dim3 grid(1024), block(256);
    entmax_attn_kernel<<<grid, block, 0, stream>>>(Q, K, V, O);
}
